// GATNodeRecommendation_55946243998128
// MI455X (gfx1250) — compile-verified, matrix operand order checked by reading, not run on hardware
//
#include <hip/hip_runtime.h>
#include <hip/hip_bf16.h>

#define N_NODES 100000
#define N_EDGES 1600000
#define TOT_E   (N_EDGES + N_NODES)   // with self loops
#define IN_CH   64
#define OUT_CH  32
#define HEADS   4
#define HC      (HEADS * OUT_CH)      // 128
#define NEG_SLOPE 0.2f

typedef __attribute__((ext_vector_type(16))) _Float16 v16h;
typedef __attribute__((ext_vector_type(8)))  float    v8f;

// ---------- monotone float <-> uint mapping for atomicMax over floats ----------
__device__ __forceinline__ unsigned fflip(float f) {
    unsigned u = __float_as_uint(f);
    return (u & 0x80000000u) ? ~u : (u | 0x80000000u);
}
__device__ __forceinline__ float funflip(unsigned u) {
    return __uint_as_float((u & 0x80000000u) ? (u & 0x7FFFFFFFu) : ~u);
}

__device__ __forceinline__ void edge_sd(const int* __restrict__ ei, int e, int& s, int& d) {
    if (e < N_EDGES) { s = ei[e]; d = ei[N_EDGES + e]; }
    else             { s = d = e - N_EDGES; }          // self loop
}

// =============================================================================
// Kernel 1: h = x @ W  via WMMA f16->f32.  One block = one 16-row M tile,
// 8 waves cover the 8 N-tiles of the 128 output columns.  K=64 -> 2 WMMAs.
// =============================================================================
__global__ __launch_bounds__(256) void k_gemm(const float* __restrict__ x,
                                              const float* __restrict__ W,
                                              float* __restrict__ h) {
    __shared__ float xs[16 * IN_CH];              // 4 KB x-tile
    const int tid = threadIdx.x;
    const int mt  = blockIdx.x;                   // 100000/16 = 6250 exact
    for (int i = tid; i < 16 * IN_CH; i += 256) {
        int r = i >> 6, c = i & 63;
        xs[i] = x[(size_t)(mt * 16 + r) * IN_CH + c];
    }
    __syncthreads();

    const int wave = tid >> 5;                    // N-tile 0..7
    const int lane = tid & 31;
    const int mrow = lane & 15;
    const int khalf = (lane >> 4) << 3;           // 0 or 8 (CDNA5 16-bit A/B lane split)
    const int ncol = wave * 16 + (lane & 15);

    v16h a0, a1, b0, b1;
    #pragma unroll
    for (int i = 0; i < 16; ++i) {
        int k = i + (i & 8) + khalf;              // documented K swizzle for 16-bit frags
        a0[i] = (_Float16)xs[mrow * IN_CH + k];
        a1[i] = (_Float16)xs[mrow * IN_CH + k + 32];
        b0[i] = (_Float16)W[(size_t)k * HC + ncol];
        b1[i] = (_Float16)W[(size_t)(k + 32) * HC + ncol];
    }
    v8f c = {};
    c = __builtin_amdgcn_wmma_f32_16x16x32_f16(false, a0, false, b0, (short)0, c, false, false);
    c = __builtin_amdgcn_wmma_f32_16x16x32_f16(false, a1, false, b1, (short)0, c, false, false);

    #pragma unroll
    for (int r = 0; r < 8; ++r) {
        int m = mt * 16 + r + khalf;              // D layout: VGPR r, lanes 16-31 -> M+8
        h[(size_t)m * HC + ncol] = c[r];
    }
}

// =============================================================================
// Kernel 2: per-node attention halves  a_src/a_dst[n][hh] = <h[n,hh,:], att>
// =============================================================================
__global__ __launch_bounds__(256) void k_att(const float* __restrict__ h,
                                             const float* __restrict__ att_src,
                                             const float* __restrict__ att_dst,
                                             float* __restrict__ a_src,
                                             float* __restrict__ a_dst) {
    int idx = blockIdx.x * blockDim.x + threadIdx.x;   // node*4 + head
    if (idx >= N_NODES * HEADS) return;
    int n = idx >> 2, hh = idx & 3;
    const float* hp = h + (size_t)n * HC + hh * OUT_CH;
    float s = 0.f, d = 0.f;
    #pragma unroll
    for (int c = 0; c < OUT_CH; ++c) {
        float v = hp[c];
        s += v * att_src[hh * OUT_CH + c];
        d += v * att_dst[hh * OUT_CH + c];
    }
    a_src[idx] = s; a_dst[idx] = d;
}

// =============================================================================
// Kernel 3: init scratch + out accumulator (harness poisons them with 0xAA)
// =============================================================================
__global__ __launch_bounds__(256) void k_init(unsigned* __restrict__ mflip,
                                              float* __restrict__ denom,
                                              float* __restrict__ out) {
    const int stride = gridDim.x * blockDim.x;
    for (int i = blockIdx.x * blockDim.x + threadIdx.x; i < N_NODES * HC; i += stride)
        out[i] = 0.f;
    for (int i = blockIdx.x * blockDim.x + threadIdx.x; i < N_NODES * HEADS; i += stride) {
        mflip[i] = 0u;                 // smallest uint, below fflip(-inf)
        denom[i] = 0.f;
    }
}

// =============================================================================
// Kernel 4: segment max of leaky-relu edge scores over dst (uint-flip atomicMax)
// =============================================================================
__global__ __launch_bounds__(256) void k_max(const int* __restrict__ ei,
                                             const float* __restrict__ a_src,
                                             const float* __restrict__ a_dst,
                                             unsigned* __restrict__ mflip) {
    int e = blockIdx.x * blockDim.x + threadIdx.x;
    if (e >= TOT_E) return;
    int s, d; edge_sd(ei, e, s, d);
    #pragma unroll
    for (int hh = 0; hh < HEADS; ++hh) {
        float v = a_src[s * HEADS + hh] + a_dst[d * HEADS + hh];
        v = v > 0.f ? v : NEG_SLOPE * v;
        atomicMax(&mflip[d * HEADS + hh], fflip(v));
    }
}

// =============================================================================
// Kernel 5: denom[dst][hh] += exp(e - m[dst][hh])
// =============================================================================
__global__ __launch_bounds__(256) void k_denom(const int* __restrict__ ei,
                                               const float* __restrict__ a_src,
                                               const float* __restrict__ a_dst,
                                               const unsigned* __restrict__ mflip,
                                               float* __restrict__ denom) {
    int e = blockIdx.x * blockDim.x + threadIdx.x;
    if (e >= TOT_E) return;
    int s, d; edge_sd(ei, e, s, d);
    #pragma unroll
    for (int hh = 0; hh < HEADS; ++hh) {
        float v = a_src[s * HEADS + hh] + a_dst[d * HEADS + hh];
        v = v > 0.f ? v : NEG_SLOPE * v;
        float m = funflip(mflip[d * HEADS + hh]);   // always finite (self loops)
        atomicAdd(&denom[d * HEADS + hh], __expf(v - m));
    }
}

// =============================================================================
// Kernel 6: scatter-aggregate  out[dst] += alpha * h[src].   One wave32 per
// edge: lane l owns channels 4l..4l+3 (all in head l>>3); float4 gather of
// h[src] + 4 coalesced f32 atomics into out[dst].  L2-resident working set.
// =============================================================================
__global__ __launch_bounds__(256) void k_agg(const int* __restrict__ ei,
                                             const float* __restrict__ a_src,
                                             const float* __restrict__ a_dst,
                                             const unsigned* __restrict__ mflip,
                                             const float* __restrict__ denom,
                                             const float* __restrict__ h,
                                             float* __restrict__ out) {
    const int lane  = threadIdx.x & 31;
    const int hh    = lane >> 3;                       // head for this lane's 4 channels
    const int wglob = (blockIdx.x * blockDim.x + threadIdx.x) >> 5;
    const int nwav  = (gridDim.x * blockDim.x) >> 5;

    for (int e = wglob; e < TOT_E; e += nwav) {
        int s, d; edge_sd(ei, e, s, d);

        // prefetch next edge's source row (global_prefetch_b8)
        int en = e + nwav;
        if (en < TOT_E) {
            int sn = (en < N_EDGES) ? ei[en] : (en - N_EDGES);
            __builtin_prefetch(h + (size_t)sn * HC + lane * 4, 0, 0);
        }

        float v = a_src[s * HEADS + hh] + a_dst[d * HEADS + hh];
        v = v > 0.f ? v : NEG_SLOPE * v;
        float m     = funflip(mflip[d * HEADS + hh]);
        float alpha = __expf(v - m) / (denom[d * HEADS + hh] + 1e-16f);

        float4 hv = ((const float4*)(h + (size_t)s * HC))[lane];
        float* op = out + (size_t)d * HC + lane * 4;
        atomicAdd(op + 0, hv.x * alpha);
        atomicAdd(op + 1, hv.y * alpha);
        atomicAdd(op + 2, hv.z * alpha);
        atomicAdd(op + 3, hv.w * alpha);
    }
}

// =============================================================================
// Kernel 7: out += bias, then in-place log_softmax over the 128 channels.
// One wave32 per node, 4 channels per lane, shfl_xor tree reductions.
// =============================================================================
__global__ __launch_bounds__(256) void k_lsm(float* __restrict__ out,
                                             const float* __restrict__ bias) {
    const int lane = threadIdx.x & 31;
    const int node = (blockIdx.x * blockDim.x + threadIdx.x) >> 5;
    if (node >= N_NODES) return;
    float* row = out + (size_t)node * HC;
    float4 v = ((float4*)row)[lane];
    float4 b = ((const float4*)bias)[lane];
    v.x += b.x; v.y += b.y; v.z += b.z; v.w += b.w;

    float mx = fmaxf(fmaxf(v.x, v.y), fmaxf(v.z, v.w));
    #pragma unroll
    for (int o = 16; o >= 1; o >>= 1) mx = fmaxf(mx, __shfl_xor(mx, o, 32));

    float se = __expf(v.x - mx) + __expf(v.y - mx) + __expf(v.z - mx) + __expf(v.w - mx);
    #pragma unroll
    for (int o = 16; o >= 1; o >>= 1) se += __shfl_xor(se, o, 32);

    float lse = mx + __logf(se);
    v.x -= lse; v.y -= lse; v.z -= lse; v.w -= lse;
    ((float4*)row)[lane] = v;
}

// =============================================================================
extern "C" void kernel_launch(void* const* d_in, const int* in_sizes, int n_in,
                              void* d_out, int out_size, void* d_ws, size_t ws_size,
                              hipStream_t stream) {
    const float* x       = (const float*)d_in[0];
    const int*   ei      = (const int*)  d_in[1];
    const float* W       = (const float*)d_in[2];
    const float* att_src = (const float*)d_in[3];
    const float* att_dst = (const float*)d_in[4];
    const float* bias    = (const float*)d_in[5];
    float*       out     = (float*)d_out;

    // scratch layout (≈56 MB)
    float*    h     = (float*)d_ws;                       // N_NODES*HC
    float*    asrc  = h    + (size_t)N_NODES * HC;        // N_NODES*HEADS
    float*    adst  = asrc + (size_t)N_NODES * HEADS;
    unsigned* mflip = (unsigned*)(adst + (size_t)N_NODES * HEADS);
    float*    denom = (float*)(mflip + (size_t)N_NODES * HEADS);

    k_gemm<<<N_NODES / 16, 256, 0, stream>>>(x, W, h);
    k_att <<<(N_NODES * HEADS + 255) / 256, 256, 0, stream>>>(h, att_src, att_dst, asrc, adst);
    k_init<<<4096, 256, 0, stream>>>(mflip, denom, out);
    k_max <<<(TOT_E + 255) / 256, 256, 0, stream>>>(ei, asrc, adst, mflip);
    k_denom<<<(TOT_E + 255) / 256, 256, 0, stream>>>(ei, asrc, adst, mflip, denom);
    k_agg <<<12288, 256, 0, stream>>>(ei, asrc, adst, mflip, denom, h, out);
    k_lsm <<<(N_NODES + 7) / 8, 256, 0, stream>>>(out, bias);
}
